// FP8Linear_89515708383266
// MI455X (gfx1250) — 8-bit matrix operand order repaired and checked by reading, compiled, not run on hardware
//
#include <hip/hip_runtime.h>
#include <hip/hip_bf16.h>

typedef __attribute__((ext_vector_type(16))) int   v16i;
typedef __attribute__((ext_vector_type(4)))  int   v4i;
typedef __attribute__((ext_vector_type(8)))  float v8f;

typedef __attribute__((address_space(1))) v4i* gas_v4i_p;
typedef __attribute__((address_space(3))) v4i* las_v4i_p;

#define FP8_MAX 448.0f
#define QBLOCK  128

#define FRAG_BYTES   2048
#define LANE_STRIDE  80
#define TILE_LDS     (32 * LANE_STRIDE)
#define BUF_LDS      (4 * TILE_LDS)
#define LDS_TOTAL    (2 * BUF_LDS)

__device__ __forceinline__ unsigned char sw_f32_to_e4m3(float f) {
    unsigned sign = (__float_as_uint(f) >> 31) << 7;
    float a = fabsf(f);
    if (a > FP8_MAX) a = FP8_MAX;
    if (a < 0.0009765625f) return (unsigned char)sign;
    int e;
    float m = frexpf(a, &e);
    (void)m;
    int E = e - 1;
    if (E < -6) {
        int mi = (int)rintf(a * 512.0f);
        if (mi > 7) mi = 7;
        return (unsigned char)(sign | mi);
    }
    int mi = (int)rintf(ldexpf(a, 3 - E));
    if (mi == 16) { mi = 8; E += 1; }
    if (E > 8 || (E == 8 && mi > 14)) return (unsigned char)(sign | 0x7E);
    return (unsigned char)(sign | ((unsigned)(E + 7) << 3) | (unsigned)(mi - 8));
}

__device__ __forceinline__ unsigned pack_fp8x2_lo(float a, float b, unsigned old) {
#if __has_builtin(__builtin_amdgcn_cvt_pk_fp8_f32)
    return (unsigned)__builtin_amdgcn_cvt_pk_fp8_f32(a, b, (int)old, false);
#else
    return (old & 0xFFFF0000u) | (unsigned)sw_f32_to_e4m3(a) |
           ((unsigned)sw_f32_to_e4m3(b) << 8);
#endif
}
__device__ __forceinline__ unsigned pack_fp8x2_hi(float a, float b, unsigned old) {
#if __has_builtin(__builtin_amdgcn_cvt_pk_fp8_f32)
    return (unsigned)__builtin_amdgcn_cvt_pk_fp8_f32(a, b, (int)old, true);
#else
    return (old & 0x0000FFFFu) | ((unsigned)sw_f32_to_e4m3(a) << 16) |
           ((unsigned)sw_f32_to_e4m3(b) << 24);
#endif
}
__device__ __forceinline__ unsigned pack_fp8x4(float a, float b, float c, float d) {
    return pack_fp8x2_hi(c, d, pack_fp8x2_lo(a, b, 0u));
}

#if __has_builtin(__builtin_amdgcn_global_load_async_to_lds_b128)
#define HAVE_ASYNC_LDS 1
#endif

__device__ __forceinline__ void wait_async0() {
#if __has_builtin(__builtin_amdgcn_s_wait_asynccnt)
    __builtin_amdgcn_s_wait_asynccnt(0);
#elif defined(HAVE_ASYNC_LDS)
    asm volatile("s_wait_asynccnt 0x0" ::: "memory");
#endif
}

__device__ __forceinline__ void copy16_g2lds(const unsigned char* g, unsigned char* l) {
#ifdef HAVE_ASYNC_LDS
    __builtin_amdgcn_global_load_async_to_lds_b128(
        (gas_v4i_p)g, (las_v4i_p)l, 0, 0);
#else
    *(int4*)l = *(const int4*)g;
#endif
}

__global__ __launch_bounds__(256)
void quant_x_kernel(const float* __restrict__ x, unsigned char* __restrict__ xq_sw,
                    float* __restrict__ xs_t, int M, int K) {
    const int Kb    = K / QBLOCK;
    const int lane  = threadIdx.x & 31;
    const int gwave = (blockIdx.x * (blockDim.x >> 5)) + (threadIdx.x >> 5);
    if (gwave >= M * Kb) return;
    const int m  = gwave / Kb;
    const int kb = gwave - m * Kb;

    const float4 v = *(const float4*)(x + (size_t)m * K + kb * QBLOCK + lane * 4);
    float amax = fmaxf(fmaxf(fabsf(v.x), fabsf(v.y)), fmaxf(fabsf(v.z), fabsf(v.w)));
    #pragma unroll
    for (int off = 16; off > 0; off >>= 1)
        amax = fmaxf(amax, __shfl_xor(amax, off, 32));

    const float s = fmaxf(amax * (1.0f / FP8_MAX), 1e-12f);
    if (lane == 0) xs_t[(size_t)kb * M + m] = s;

    const float q0 = fminf(fmaxf(v.x / s, -FP8_MAX), FP8_MAX);
    const float q1 = fminf(fmaxf(v.y / s, -FP8_MAX), FP8_MAX);
    const float q2 = fminf(fmaxf(v.z / s, -FP8_MAX), FP8_MAX);
    const float q3 = fminf(fmaxf(v.w / s, -FP8_MAX), FP8_MAX);
    const unsigned packed = pack_fp8x4(q0, q1, q2, q3);

    const int mt  = m >> 4, r = m & 15;
    const int blk = lane >> 4;
    const int p   = ((lane >> 3) & 1) * 2 + ((lane >> 1) & 1);
    const int h   = (lane >> 2) & 1;
    const int idx = blk * 8 + p * 2 + (lane & 1);
    const int dl  = r + 16 * h;
    const size_t frag = (size_t)mt * Kb + kb;
    *(unsigned*)(xq_sw + frag * FRAG_BYTES + dl * 64 + idx * 4) = packed;
}

__global__ __launch_bounds__(256)
void quant_w_kernel(const float* __restrict__ wq_f32, unsigned char* __restrict__ wq_sw,
                    int N, int K) {
    const int Kb = K / QBLOCK;
    const size_t i = (size_t)blockIdx.x * blockDim.x + threadIdx.x;
    if (i >= (size_t)N * K / 4) return;
    const int n  = (int)(i / (K / 4));
    const int k  = (int)(i % (K / 4)) * 4;

    const float4 v = *(const float4*)(wq_f32 + (size_t)n * K + k);
    const unsigned packed = pack_fp8x4(v.x, v.y, v.z, v.w);

    const int nt  = n >> 4, l16 = n & 15;
    const int kb  = k >> 7, kk = k & 127;
    const int c   = kk >> 5;
    const int h   = (kk >> 4) & 1;
    const int dl  = l16 + 16 * h;
    const int dwd = c * 4 + ((kk & 15) >> 2);
    const size_t frag = (size_t)nt * Kb + kb;
    *(unsigned*)(wq_sw + frag * FRAG_BYTES + dl * 64 + dwd * 4) = packed;
}

__global__ __launch_bounds__(256)
void fp8_gemm_kernel(const unsigned char* __restrict__ Aq,
                     const float*         __restrict__ AsT,
                     const unsigned char* __restrict__ Bq,
                     const float*         __restrict__ Bs,
                     float*               __restrict__ C,
                     int M, int N, int K) {
    __shared__ __align__(16) unsigned char smem[LDS_TOTAL];

    const int lane = threadIdx.x & 31;
    const int wave = threadIdx.x >> 5;
    const int h    = lane >> 4;
    const int l16  = lane & 15;
    const int Kb   = K / QBLOCK;

    const int mt0     = blockIdx.y * 16 + wave;
    const int nt0     = blockIdx.x * 4;
    const int n_base  = nt0 * 16;
    const int nb      = n_base >> 7;

    auto stage_b = [&](int kb, int buf) {
        #pragma unroll
        for (int rr = 0; rr < 2; ++rr) {
            const int e  = (int)threadIdx.x + rr * 256;
            const int t  = e >> 7;
            const int ei = e & 127;
            const unsigned char* g =
                Bq + ((size_t)(nt0 + t) * Kb + kb) * FRAG_BYTES + ei * 16;
            unsigned char* l =
                smem + buf * BUF_LDS + t * TILE_LDS + (ei >> 2) * LANE_STRIDE + (ei & 3) * 16;
            copy16_g2lds(g, l);
        }
    };

    v8f acc[2][4];
    #pragma unroll
    for (int u = 0; u < 2; ++u)
        #pragma unroll
        for (int t = 0; t < 4; ++t)
            acc[u][t] = (v8f){};

    stage_b(0, 0);
    int buf = 0;

    for (int kb = 0; kb < Kb; ++kb) {
        wait_async0();
        __syncthreads();
        if (kb + 1 < Kb) stage_b(kb + 1, buf ^ 1);

        v16i a[2];
        #pragma unroll
        for (int u = 0; u < 2; ++u) {
            const int mt = mt0 + u * 8;
            const unsigned char* ap =
                Aq + ((size_t)mt * Kb + kb) * FRAG_BYTES + lane * 64;
            #pragma unroll
            for (int c = 0; c < 4; ++c) {
                const int4 d = *(const int4*)(ap + c * 16);
                a[u][c * 4 + 0] = d.x; a[u][c * 4 + 1] = d.y;
                a[u][c * 4 + 2] = d.z; a[u][c * 4 + 3] = d.w;
            }
        }

        float scl[2][8];
        {
            const float ws = Bs[(size_t)nb * Kb + kb];
            #pragma unroll
            for (int u = 0; u < 2; ++u) {
                const int m_base = (mt0 + u * 8) * 16;
                const float* xsp = AsT + (size_t)kb * M + m_base + 8 * h;
                const float4 x0 = *(const float4*)(xsp);
                const float4 x1 = *(const float4*)(xsp + 4);
                scl[u][0] = x0.x * ws; scl[u][1] = x0.y * ws;
                scl[u][2] = x0.z * ws; scl[u][3] = x0.w * ws;
                scl[u][4] = x1.x * ws; scl[u][5] = x1.y * ws;
                scl[u][6] = x1.z * ws; scl[u][7] = x1.w * ws;
            }
        }

        const unsigned char* lbase = smem + buf * BUF_LDS + lane * LANE_STRIDE;
        #pragma unroll
        for (int t = 0; t < 4; ++t) {
            v16i b;
            #pragma unroll
            for (int c = 0; c < 4; ++c) {
                const int4 d = *(const int4*)(lbase + t * TILE_LDS + c * 16);
                b[c * 4 + 0] = d.x; b[c * 4 + 1] = d.y;
                b[c * 4 + 2] = d.z; b[c * 4 + 3] = d.w;
            }
            #pragma unroll
            for (int u = 0; u < 2; ++u) {
                v8f zero = {};
                v8f part = __builtin_amdgcn_wmma_f32_16x16x128_fp8_fp8(
                    a[u], b, (short)0, zero, false, false);
                #pragma unroll
                for (int i = 0; i < 8; ++i)
                    acc[u][t][i] += part[i] * scl[u][i];
            }
        }
        buf ^= 1;
    }

    #pragma unroll
    for (int u = 0; u < 2; ++u) {
        const int m_base = (mt0 + u * 8) * 16;
        #pragma unroll
        for (int t = 0; t < 4; ++t) {
            const int n0 = n_base + t * 16;
            #pragma unroll
            for (int i = 0; i < 8; ++i)
                C[(size_t)(m_base + i + 8 * h) * N + n0 + l16] = acc[u][t][i];
        }
    }
}

extern "C" void kernel_launch(void* const* d_in, const int* in_sizes, int n_in,
                              void* d_out, int out_size, void* d_ws, size_t ws_size,
                              hipStream_t stream) {
    (void)in_sizes; (void)n_in; (void)out_size; (void)ws_size;
    const float* x   = (const float*)d_in[0];
    const float* w_q = (const float*)d_in[1];
    const float* w_s = (const float*)d_in[2];
    float* out = (float*)d_out;

    const int M = 4096, K = 2048, N = 8192;
    const int Kb = K / QBLOCK;

    unsigned char* xq_sw = (unsigned char*)d_ws;
    unsigned char* wq_sw = xq_sw + (size_t)M * K;
    float*         xs_t  = (float*)(wq_sw + (size_t)N * K);

    {
        int nwaves = M * Kb;
        quant_x_kernel<<<(nwaves + 7) / 8, 256, 0, stream>>>(x, xq_sw, xs_t, M, K);
    }
    {
        size_t n4 = (size_t)N * K / 4;
        quant_w_kernel<<<(unsigned)((n4 + 255) / 256), 256, 0, stream>>>(w_q, wq_sw, N, K);
    }
    {
        dim3 grid(N / 64, M / 256);
        fp8_gemm_kernel<<<grid, 256, 0, stream>>>(xq_sw, xs_t, wq_sw, w_s, out, M, N, K);
    }
}
